// TimeDependentAttention_48430051229898
// MI455X (gfx1250) — matrix operand order repaired and checked by reading, not compiled, not run on hardware
//
#include <hip/hip_runtime.h>
#include <hip/hip_bf16.h>

// ---------------------------------------------------------------------------
// Time-dependent attention for gfx1250 (MI455X), wave32 + WMMA bf16.
//   B=8, L=1024, D=768, H=12, Dh=64, Dt=256
// Pipeline:
//   1) cvt x, Wx_q/k/v, W_o -> bf16 (workspace)
//   2) tshift[s][b][:] = t @ Wt_s^T  (f32, tiny)
//   3) Q/K/V = Xbf @ Wbf^T + tshift    (WMMA gemm, 2-deep ping-pong pipeline)
//   4) flash attention per (b,h,64 q rows): WMMA QK^T + online softmax + WMMA PV
//      (K tile staged to LDS via GLOBAL_LOAD_ASYNC_TO_LDS when available)
//   5) out = Obf @ Wo^T + b_o          (WMMA gemm, f32 out)
// ---------------------------------------------------------------------------

typedef __bf16 bf16;
typedef __attribute__((ext_vector_type(8)))  __bf16 v8bf;
typedef __attribute__((ext_vector_type(16))) __bf16 v16bf;
typedef __attribute__((ext_vector_type(8)))  float  v8f;
typedef __attribute__((vector_size(4 * sizeof(int)))) int v4i_t;  // async-copy unit

#define D_MODEL 768
#define SEQ     1024
#define BATCH   8
#define NH      12
#define HD      64
#define DT      256

#if defined(__HIP_DEVICE_COMPILE__) &&                                       \
    __has_builtin(__builtin_amdgcn_global_load_async_to_lds_b128) &&         \
    __has_builtin(__builtin_amdgcn_s_wait_asynccnt)
#define HAS_ASYNC_LDS 1
#else
#define HAS_ASYNC_LDS 0
#endif

__device__ __forceinline__ v8f wmma_bf16(v16bf a, v16bf b, v8f c) {
  // (neg_a, A, neg_b, B, c_mod, C, reuse_a, reuse_b)
  return __builtin_amdgcn_wmma_f32_16x16x32_bf16(false, a, false, b, (short)0, c,
                                                 false, false);
}

// A-fragment style: two contiguous 8-half (16B) runs per lane.
__device__ __forceinline__ v16bf frag2(const bf16* p0, const bf16* p1) {
  v8bf a = *(const v8bf*)p0;
  v8bf b = *(const v8bf*)p1;
  v16bf r;
#pragma unroll
  for (int i = 0; i < 8; ++i) { r[i] = a[i]; r[i + 8] = b[i]; }
  return r;
}

// B-fragment style: 16 contiguous halves (32B) per lane.
__device__ __forceinline__ v16bf frag16(const bf16* p) { return frag2(p, p + 8); }

// reductions across a 16-lane half of the wave32
__device__ __forceinline__ float redmax16(float v) {
#pragma unroll
  for (int m = 1; m <= 8; m <<= 1) v = fmaxf(v, __shfl_xor(v, m, 32));
  return v;
}
__device__ __forceinline__ float redsum16(float v) {
#pragma unroll
  for (int m = 1; m <= 8; m <<= 1) v += __shfl_xor(v, m, 32);
  return v;
}

// ---------------------------------------------------------------------------
__global__ void cvt_bf16_kernel(const float* __restrict__ src,
                                bf16* __restrict__ dst, int n) {
  int i = blockIdx.x * blockDim.x + threadIdx.x;
  if (i < n) dst[i] = (bf16)src[i];
}

// tshift[s*B*D + b*D + j] = sum_k t[b][k] * Wt_s[j][k]
__global__ void tshift_kernel(const float* __restrict__ t,
                              const float* __restrict__ Wq,
                              const float* __restrict__ Wk,
                              const float* __restrict__ Wv,
                              float* __restrict__ out) {
  int tid = blockIdx.x * blockDim.x + threadIdx.x;
  if (tid >= 3 * BATCH * D_MODEL) return;
  int s = tid / (BATCH * D_MODEL);
  int r = tid % (BATCH * D_MODEL);
  int b = r / D_MODEL, j = r % D_MODEL;
  const float* W  = (s == 0) ? Wq : (s == 1) ? Wk : Wv;
  const float* tv = t + b * DT;
  const float* wv = W + (size_t)j * DT;
  float acc = 0.f;
#pragma unroll 4
  for (int k = 0; k < DT; ++k) acc += tv[k] * wv[k];
  out[tid] = acc;
}

// ---------------------------------------------------------------------------
// C[m][n] = sum_k A[m][k] * W[n][k]  (+ shift), M = B*L, N = K = 768.
// 128 threads = 4 waves; each wave computes a 16x64 tile with WMMA.
// k-loop unrolled by 2: two fragment buffer sets alternate load/compute roles
// by SSA renaming (no register-rotation moves), loads overlap WMMAs.
// mode 0: shift = tshift per batch, bf16 output in split-head [B,H,L,64]
// mode 1: shift = b_o per column,   f32 output [M, 768]
__global__ __launch_bounds__(128)
void gemm_bf16_kernel(const bf16* __restrict__ A, const bf16* __restrict__ W,
                      const float* __restrict__ shift, int mode,
                      bf16* __restrict__ outBf, float* __restrict__ outF) {
  const int K = D_MODEL;
  int wave = threadIdx.x >> 5;
  int lane = threadIdx.x & 31;
  int lh = lane >> 4, l15 = lane & 15;
  int mBase = (int)blockIdx.x * 64 + wave * 16;
  int nBase = (int)blockIdx.y * 64;

  v8f acc[4];
#pragma unroll
  for (int c = 0; c < 4; ++c)
#pragma unroll
    for (int i = 0; i < 8; ++i) acc[c][i] = 0.f;

  const bf16* arow = A + (size_t)(mBase + l15) * K;
  const bf16* wrow[4];
#pragma unroll
  for (int c = 0; c < 4; ++c)
    wrow[c] = W + (size_t)(nBase + c * 16 + l15) * K + 16 * lh;

  // prologue: fragment set 0 holds k = 0
  v16bf af0 = frag2(arow + 16 * lh, arow + 8 + 16 * lh);
  v16bf bf0[4];
#pragma unroll
  for (int c = 0; c < 4; ++c) bf0[c] = frag16(wrow[c]);
  v16bf af1 = af0;
  v16bf bf1[4];
#pragma unroll
  for (int c = 0; c < 4; ++c) bf1[c] = bf0[c];

  // 12 iterations of a 64-wide k step (K = 768)
  for (int k0 = 0; k0 < K; k0 += 64) {
    // load set 1 (k0+32): always in range (max 736 < 768)
    af1 = frag2(arow + k0 + 32 + 16 * lh, arow + k0 + 40 + 16 * lh);
#pragma unroll
    for (int c = 0; c < 4; ++c) bf1[c] = frag16(wrow[c] + k0 + 32);
    __builtin_prefetch(arow + k0 + 64, 0, 3);
    __builtin_prefetch(wrow[0] + k0 + 64, 0, 3);
    // compute with set 0 (k0)
#pragma unroll
    for (int c = 0; c < 4; ++c) acc[c] = wmma_bf16(af0, bf0[c], acc[c]);
    // reload set 0 with k0+64 (dead after the WMMAs above; no copies needed)
    if (k0 + 64 < K) {
      af0 = frag2(arow + k0 + 64 + 16 * lh, arow + k0 + 72 + 16 * lh);
#pragma unroll
      for (int c = 0; c < 4; ++c) bf0[c] = frag16(wrow[c] + k0 + 64);
    }
    // compute with set 1 (k0+32)
#pragma unroll
    for (int c = 0; c < 4; ++c) acc[c] = wmma_bf16(af1, bf1[c], acc[c]);
  }

#pragma unroll
  for (int c = 0; c < 4; ++c) {
    int n = nBase + c * 16 + l15;
#pragma unroll
    for (int g = 0; g < 8; ++g) {
      int m = mBase + g + 8 * lh;  // C layout: row = g + 8*lanehalf
      float v = acc[c][g];
      if (mode == 0) {
        v += shift[(m >> 10) * D_MODEL + n];
        int b = m >> 10, l = m & (SEQ - 1), hh = n >> 6, dh = n & (HD - 1);
        outBf[(((size_t)(b * NH + hh) * SEQ) + l) * HD + dh] = (bf16)v;
      } else {
        v += shift[n];
        outF[(size_t)m * D_MODEL + n] = v;
      }
    }
  }
}

// ---------------------------------------------------------------------------
// Flash attention. grid.x = B*H*(L/64). 128 threads = 4 waves; wave w owns
// 16 query rows. Key blocks of 32; K tile row-major in LDS (async copy when
// available), V tile transposed through registers.
__global__ __launch_bounds__(128)
void attn_kernel(const bf16* __restrict__ Q, const bf16* __restrict__ K,
                 const bf16* __restrict__ V, const float* __restrict__ bias,
                 bf16* __restrict__ O) {
  __shared__ __align__(16) bf16 sK[32][72];       // 32 keys x 64 dims (+pad)
  __shared__ __align__(16) bf16 sVt[64][40];      // transposed: [dim][key]
  __shared__ __align__(16) bf16 sP[4][16][40];    // wave-private P tiles

  int bh = blockIdx.x >> 4;  // b*12 + h
  int qb = blockIdx.x & 15;
  int h = bh % NH;
  int wave = threadIdx.x >> 5;
  int lane = threadIdx.x & 31;
  int lh = lane >> 4, l15 = lane & 15;
  int qBase = qb * 64 + wave * 16;

  const bf16* Qb = Q + (size_t)bh * SEQ * HD;
  const bf16* Kb = K + (size_t)bh * SEQ * HD;
  const bf16* Vb = V + (size_t)bh * SEQ * HD;

  // Q fragments (d=64 -> two 16x32 A fragments), kept in registers
  const bf16* qrow = Qb + (size_t)(qBase + l15) * HD;
  v16bf qa[2];
#pragma unroll
  for (int kc = 0; kc < 2; ++kc)
    qa[kc] = frag2(qrow + kc * 32 + 16 * lh, qrow + kc * 32 + 8 + 16 * lh);

  v8f o[4];
  float mrow[8], lrow[8];
#pragma unroll
  for (int c = 0; c < 4; ++c)
#pragma unroll
    for (int i = 0; i < 8; ++i) o[c][i] = 0.f;
#pragma unroll
  for (int g = 0; g < 8; ++g) { mrow[g] = -1e30f; lrow[g] = 0.f; }

  for (int kb = 0; kb < 32; ++kb) {
    __syncthreads();  // protect tiles from previous iteration
    // cooperative load: K tile row-major (async->LDS if available),
    // V tile transposed into LDS through registers
#pragma unroll
    for (int p = 0; p < 2; ++p) {
      int idx = p * 128 + (int)threadIdx.x;  // 0..255 -> (row, 8-elem segment)
      int r = idx >> 3, seg = idx & 7;
#if HAS_ASYNC_LDS
      __builtin_amdgcn_global_load_async_to_lds_b128(
          (v4i_t*)(Kb + (size_t)(kb * 32 + r) * HD + seg * 8),
          (v4i_t*)&sK[r][seg * 8], 0, 0);
#else
      *(v8bf*)&sK[r][seg * 8] =
          *(const v8bf*)(Kb + (size_t)(kb * 32 + r) * HD + seg * 8);
#endif
      v8bf vv = *(const v8bf*)(Vb + (size_t)(kb * 32 + r) * HD + seg * 8);
#pragma unroll
      for (int i = 0; i < 8; ++i) sVt[seg * 8 + i][r] = vv[i];
    }
#if HAS_ASYNC_LDS
    __builtin_amdgcn_s_wait_asynccnt(0);  // barrier does not cover ASYNCcnt
#endif
    __syncthreads();

    // S = Q * K^T for this wave's 16 rows x 32 keys (two 16-col halves)
    v8f s[2];
#pragma unroll
    for (int nh = 0; nh < 2; ++nh) {
#pragma unroll
      for (int i = 0; i < 8; ++i) s[nh][i] = 0.f;
#pragma unroll
      for (int kc = 0; kc < 2; ++kc)
        s[nh] = wmma_bf16(qa[kc],
                          frag16(&sK[nh * 16 + l15][kc * 32 + 16 * lh]), s[nh]);
    }

    // scale + rel-pos bias + online softmax update
    const float* bb = bias + ((size_t)h * SEQ) * SEQ + (size_t)kb * 32;
#pragma unroll
    for (int g = 0; g < 8; ++g) {
      int row = qBase + g + 8 * lh;
      float s0 = s[0][g] * 0.125f + bb[(size_t)row * SEQ + l15];
      float s1 = s[1][g] * 0.125f + bb[(size_t)row * SEQ + 16 + l15];
      float rm = redmax16(fmaxf(s0, s1));
      float mnew = fmaxf(mrow[g], rm);
      float corr = __expf(mrow[g] - mnew);
      float p0 = __expf(s0 - mnew);
      float p1 = __expf(s1 - mnew);
      float rs = redsum16(p0 + p1);
      lrow[g] = lrow[g] * corr + rs;
      mrow[g] = mnew;
#pragma unroll
      for (int c = 0; c < 4; ++c) o[c][g] *= corr;
      int m = g + 8 * lh;  // C-layout -> P tile (row, col) in LDS
      sP[wave][m][l15]      = (bf16)p0;
      sP[wave][m][16 + l15] = (bf16)p1;
    }

    // O += P * V  (P as A fragment from LDS, V^T columns contiguous in LDS)
    v16bf pa = frag2(&sP[wave][l15][16 * lh], &sP[wave][l15][8 + 16 * lh]);
#pragma unroll
    for (int c = 0; c < 4; ++c)
      o[c] = wmma_bf16(pa, frag16(&sVt[c * 16 + l15][16 * lh]), o[c]);
  }

  // normalize and store merged-head bf16 [B, L, D]
  int b = bh / NH;
#pragma unroll
  for (int g = 0; g < 8; ++g) {
    float inv = 1.f / lrow[g];
    int row = qBase + g + 8 * lh;
    size_t rIdx = (size_t)b * SEQ + row;
#pragma unroll
    for (int c = 0; c < 4; ++c)
      O[rIdx * D_MODEL + h * HD + c * 16 + l15] = (bf16)(o[c][g] * inv);
  }
}

// ---------------------------------------------------------------------------
extern "C" void kernel_launch(void* const* d_in, const int* in_sizes, int n_in,
                              void* d_out, int out_size, void* d_ws,
                              size_t ws_size, hipStream_t stream) {
  (void)in_sizes; (void)n_in; (void)out_size; (void)ws_size;
  const float* x    = (const float*)d_in[0];
  const float* t    = (const float*)d_in[1];
  const float* bias = (const float*)d_in[2];
  const float* Wxq  = (const float*)d_in[3];
  const float* Wxk  = (const float*)d_in[4];
  const float* Wxv  = (const float*)d_in[5];
  const float* Wtq  = (const float*)d_in[6];
  const float* Wtk  = (const float*)d_in[7];
  const float* Wtv  = (const float*)d_in[8];
  const float* Wo   = (const float*)d_in[9];
  const float* bo   = (const float*)d_in[10];
  float* out = (float*)d_out;

  char* ws = (char*)d_ws;
  const size_t nX = (size_t)BATCH * SEQ * D_MODEL;  // 6,291,456
  const size_t nW = (size_t)D_MODEL * D_MODEL;      //   589,824
  size_t off = 0;
  bf16* Xb  = (bf16*)(ws + off); off += nX * 2;
  bf16* Wqb = (bf16*)(ws + off); off += nW * 2;
  bf16* Wkb = (bf16*)(ws + off); off += nW * 2;
  bf16* Wvb = (bf16*)(ws + off); off += nW * 2;
  bf16* Wob = (bf16*)(ws + off); off += nW * 2;
  bf16* Qb  = (bf16*)(ws + off); off += nX * 2;
  bf16* Kb  = (bf16*)(ws + off); off += nX * 2;
  bf16* Vb  = (bf16*)(ws + off); off += nX * 2;
  bf16* Ob  = (bf16*)(ws + off); off += nX * 2;
  float* tsh = (float*)(ws + off);  // 3 * B * D floats

  cvt_bf16_kernel<<<(int)((nX + 255) / 256), 256, 0, stream>>>(x, Xb, (int)nX);
  cvt_bf16_kernel<<<(int)((nW + 255) / 256), 256, 0, stream>>>(Wxq, Wqb, (int)nW);
  cvt_bf16_kernel<<<(int)((nW + 255) / 256), 256, 0, stream>>>(Wxk, Wkb, (int)nW);
  cvt_bf16_kernel<<<(int)((nW + 255) / 256), 256, 0, stream>>>(Wxv, Wvb, (int)nW);
  cvt_bf16_kernel<<<(int)((nW + 255) / 256), 256, 0, stream>>>(Wo,  Wob, (int)nW);
  tshift_kernel<<<(3 * BATCH * D_MODEL + 255) / 256, 256, 0, stream>>>(
      t, Wtq, Wtk, Wtv, tsh);

  dim3 gGemm(BATCH * SEQ / 64, D_MODEL / 64);
  gemm_bf16_kernel<<<gGemm, 128, 0, stream>>>(Xb, Wqb, tsh, 0, Qb, nullptr);
  gemm_bf16_kernel<<<gGemm, 128, 0, stream>>>(Xb, Wkb, tsh + BATCH * D_MODEL, 0,
                                              Kb, nullptr);
  gemm_bf16_kernel<<<gGemm, 128, 0, stream>>>(Xb, Wvb, tsh + 2 * BATCH * D_MODEL,
                                              0, Vb, nullptr);

  attn_kernel<<<BATCH * NH * (SEQ / 64), 128, 0, stream>>>(Qb, Kb, Vb, bias, Ob);

  gemm_bf16_kernel<<<gGemm, 128, 0, stream>>>(Ob, Wob, bo, 1, nullptr, out);
}
